// MILAN_65953517797949
// MI455X (gfx1250) — matrix operand order repaired and checked by reading, not compiled, not run on hardware
//
#include <hip/hip_runtime.h>

#define HIDW 128
#define NHEADS 8
#define HDIM 16

typedef __attribute__((ext_vector_type(16))) __bf16 v16bf;
typedef __attribute__((ext_vector_type(8)))  float  v8f;

__device__ __forceinline__ unsigned int f32_to_bf16_bits(float f) {
  unsigned int u = __builtin_bit_cast(unsigned int, f);
  return ((u + 0x7FFFu + ((u >> 16) & 1u)) >> 16) & 0xFFFFu;
}

__device__ __forceinline__ float gelu_exact(float x) {
  return 0.5f * x * (1.0f + erff(x * 0.70710678118654752f));
}

__device__ __forceinline__ void atomic_max_f32(float* addr, float val) {
  unsigned int* ua = reinterpret_cast<unsigned int*>(addr);
  unsigned int cur = *ua;
  while (__builtin_bit_cast(float, cur) < val) {
    unsigned int prev = atomicCAS(ua, cur, __builtin_bit_cast(unsigned int, val));
    if (prev == cur) break;
    cur = prev;
  }
}

// ---------------------------------------------------------------------------
// Generic WMMA GEMM:  Y[M,128] = concat_seg(A)[M,Ktot] @ W[128,Ktot]^T + bias
// Up to 3 K-segments, each with optional row-gather map (fuses the edge concat).
// Block tile: 32 rows x 128 cols. 8 waves, each owns a 16-col slice and two
// 16-row accumulators -> 2 WMMAs per B fragment. Row-gather pointers hoisted
// out of the K loop; A staged to LDS as bf16 via float4 (b128) loads.
// ---------------------------------------------------------------------------
__global__ __launch_bounds__(256) void wmma_gemm_kernel(
    const float* __restrict__ A0, const int* __restrict__ R0, int W0,
    const float* __restrict__ A1, const int* __restrict__ R1, int W1,
    const float* __restrict__ A2, const int* __restrict__ R2, int W2,
    const unsigned short* __restrict__ Wt,   // bf16 bits, [128, Ktot] row-major
    const float* __restrict__ bias,          // [128] or null
    float* __restrict__ Y, int M, int Ktot) {
  __shared__ unsigned short Atile[32 * 32];  // 2 KB

  const int tid  = threadIdx.x;
  const int wave = tid >> 5;
  const int lane = tid & 31;
  const int mbase = blockIdx.x << 5;   // 32 rows per block
  const int nbase = wave << 4;
  const int ml   = lane & 15;
  const int half = lane >> 4;

  // staging: each thread owns one row / 4 consecutive cols of the 32x32 tile
  const int srow = tid >> 3;           // 0..31
  const int scol = (tid & 7) * 4;      // 0,4,...,28
  const int g    = mbase + srow;
  const bool gok = g < M;

  // hoisted per-segment row pointers (rowmap read once per thread)
  const float* p0 = nullptr; const float* p1 = nullptr; const float* p2 = nullptr;
  if (gok) {
    if (W0) { long long r = R0 ? (long long)R0[g] : (long long)g; p0 = A0 + r * (long long)W0 + scol; }
    if (W1) { long long r = R1 ? (long long)R1[g] : (long long)g; p1 = A1 + r * (long long)W1 + scol; }
    if (W2) { long long r = R2 ? (long long)R2[g] : (long long)g; p2 = A2 + r * (long long)W2 + scol; }
  }

  v8f acc0 = {0.f, 0.f, 0.f, 0.f, 0.f, 0.f, 0.f, 0.f};
  v8f acc1 = {0.f, 0.f, 0.f, 0.f, 0.f, 0.f, 0.f, 0.f};

  const unsigned short* wrow = Wt + (size_t)(nbase + ml) * Ktot + 16 * half;

  for (int kk = 0; kk < Ktot; kk += 32) {
    const float* ap; int cb;
    if (kk < W0)           { ap = p0; cb = kk; }
    else if (kk < W0 + W1) { ap = p1; cb = kk - W0; }
    else                   { ap = p2; cb = kk - W0 - W1; }

    float4 f = {0.f, 0.f, 0.f, 0.f};
    if (gok) f = *reinterpret_cast<const float4*>(ap + cb);
    unsigned int pk0 = f32_to_bf16_bits(f.x) | (f32_to_bf16_bits(f.y) << 16);
    unsigned int pk1 = f32_to_bf16_bits(f.z) | (f32_to_bf16_bits(f.w) << 16);

    __syncthreads();  // previous iteration's LDS reads complete
    unsigned int* dp = reinterpret_cast<unsigned int*>(&Atile[srow * 32 + scol]);
    dp[0] = pk0;
    dp[1] = pk1;
    __syncthreads();

    // A fragments: lane holds rows ml and ml+16;
    // VGPR v -> K = 16*half + 2v (contiguous, matches B)
    union { unsigned int u[8]; v16bf v; } a0, a1;
    const unsigned int* lds32 = reinterpret_cast<const unsigned int*>(Atile);
#pragma unroll
    for (int vv = 0; vv < 8; ++vv) {
      int k0 = 16 * half + 2 * vv;
      a0.u[vv] = lds32[ml * 16 + (k0 >> 1)];
      a1.u[vv] = lds32[(16 + ml) * 16 + (k0 >> 1)];
    }

    // B fragment: lane holds output col (row of W); contiguous K=16*half..+15
    v16bf bfr = *reinterpret_cast<const v16bf*>(wrow + kk);
    if (kk + 32 < Ktot) __builtin_prefetch(wrow + kk + 32, 0, 1);

    acc0 = __builtin_amdgcn_wmma_f32_16x16x32_bf16(
        false, a0.v, false, bfr, (short)0, acc0, false, false);
    acc1 = __builtin_amdgcn_wmma_f32_16x16x32_bf16(
        false, a1.v, false, bfr, (short)0, acc1, false, false);
  }

  const int col = nbase + ml;
  const float bv = bias ? bias[col] : 0.f;
#pragma unroll
  for (int r = 0; r < 8; ++r) {
    int row = mbase + r + 8 * half;
    if (row < M)      Y[(size_t)row * HIDW + col] = acc0[r] + bv;
    int row2 = row + 16;
    if (row2 < M)     Y[(size_t)row2 * HIDW + col] = acc1[r] + bv;
  }
}

// ---------------------------------------------------------------------------
// Row LayerNorm over 128 cols (one wave per row) with fused input modes:
//   mode 0: x = A            mode 1: x = A*sigmoid(B)+C     mode 2: x = C+gelu(A)
// Optional GELU on the output.
// ---------------------------------------------------------------------------
__global__ __launch_bounds__(256) void ln_fused_kernel(
    const float* __restrict__ A, const float* __restrict__ B, const float* __restrict__ C,
    const float* __restrict__ g, const float* __restrict__ b,
    float* __restrict__ out, int R, int mode, int gelu_out) {
  int row  = blockIdx.x * 8 + (threadIdx.x >> 5);
  int lane = threadIdx.x & 31;
  if (row >= R) return;
  size_t base = (size_t)row * HIDW;
  float x[4];
#pragma unroll
  for (int j = 0; j < 4; ++j) {
    int c = lane + j * 32;
    float a = A[base + c];
    if (mode == 1)      x[j] = a * (1.0f / (1.0f + expf(-B[base + c]))) + C[base + c];
    else if (mode == 2) x[j] = C[base + c] + gelu_exact(a);
    else                x[j] = a;
  }
  float s = x[0] + x[1] + x[2] + x[3];
#pragma unroll
  for (int off = 16; off > 0; off >>= 1) s += __shfl_xor(s, off, 32);
  float mean = s * (1.0f / 128.0f);
  float q = 0.f;
#pragma unroll
  for (int j = 0; j < 4; ++j) { float d = x[j] - mean; q += d * d; }
#pragma unroll
  for (int off = 16; off > 0; off >>= 1) q += __shfl_xor(q, off, 32);
  float inv = rsqrtf(q * (1.0f / 128.0f) + 1e-5f);
#pragma unroll
  for (int j = 0; j < 4; ++j) {
    int c = lane + j * 32;
    float y = g[c] * (x[j] - mean) * inv + b[c];
    if (gelu_out) y = gelu_exact(y);
    out[base + c] = y;
  }
}

__global__ void f2bf_kernel(const float* __restrict__ s, unsigned short* __restrict__ d, int n) {
  int i = blockIdx.x * 256 + threadIdx.x;
  if (i < n) d[i] = (unsigned short)f32_to_bf16_bits(s[i]);
}

__global__ void fill_kernel(float* p, float v, long long n) {
  long long i = (long long)blockIdx.x * 256 + threadIdx.x;
  if (i < n) p[i] = v;
}

__global__ void add2_kernel(const float* __restrict__ a, const float* __restrict__ b,
                            float* __restrict__ y, long long n) {
  long long i = (long long)blockIdx.x * 256 + threadIdx.x;
  if (i < n) y[i] = a[i] + b[i];
}

__global__ void elu1_kernel(float* p, long long n) {
  long long i = (long long)blockIdx.x * 256 + threadIdx.x;
  if (i < n) { float x = p[i]; p[i] = (x > 0.f ? x : expm1f(x)) + 1.0f; }
}

// ---- segment-softmax attention over dst ----
__global__ __launch_bounds__(256) void attn_score_kernel(
    const float* __restrict__ q, const float* __restrict__ k, const float* __restrict__ eemb,
    const int* __restrict__ src, const int* __restrict__ dst,
    float* __restrict__ score, float* __restrict__ segmax, int E) {
  int idx = blockIdx.x * 256 + threadIdx.x;
  if (idx >= E * NHEADS) return;
  int e = idx >> 3, hh = idx & 7;
  int sn = src[e], dn = dst[e];
  const float* qp = q    + (size_t)dn * HIDW + hh * HDIM;
  const float* kp = k    + (size_t)sn * HIDW + hh * HDIM;
  const float* ep = eemb + (size_t)e  * HIDW + hh * HDIM;
  float acc = 0.f;
#pragma unroll
  for (int i = 0; i < HDIM; ++i) acc += qp[i] * (kp[i] + ep[i]);
  acc *= 0.25f;  // 1/sqrt(16)
  score[idx] = acc;
  atomic_max_f32(&segmax[(size_t)dn * NHEADS + hh], acc);
}

__global__ __launch_bounds__(256) void attn_exp_kernel(
    float* __restrict__ score, const int* __restrict__ dst,
    const float* __restrict__ segmax, float* __restrict__ segsum, int E) {
  int idx = blockIdx.x * 256 + threadIdx.x;
  if (idx >= E * NHEADS) return;
  int e = idx >> 3, hh = idx & 7;
  size_t s = (size_t)dst[e] * NHEADS + hh;
  float a = expf(score[idx] - segmax[s]);
  score[idx] = a;
  atomicAdd(&segsum[s], a);
}

__global__ __launch_bounds__(256) void attn_agg_kernel(
    const float* __restrict__ alpha, const int* __restrict__ src, const int* __restrict__ dst,
    const float* __restrict__ segsum, const float* __restrict__ v, const float* __restrict__ eemb,
    float* __restrict__ agg, int E) {
  int idx = blockIdx.x * 256 + threadIdx.x;
  if (idx >= E * NHEADS) return;
  int e = idx >> 3, hh = idx & 7;
  int sn = src[e], dn = dst[e];
  float a = alpha[idx] / (segsum[(size_t)dn * NHEADS + hh] + 1e-16f);
  const float* vp = v    + (size_t)sn * HIDW + hh * HDIM;
  const float* ep = eemb + (size_t)e  * HIDW + hh * HDIM;
  float* ap       = agg  + (size_t)dn * HIDW + hh * HDIM;
#pragma unroll
  for (int i = 0; i < HDIM; ++i) atomicAdd(&ap[i], a * (vp[i] + ep[i]));
}

// ---- GraphNorm (column stats over all nodes) ----
__global__ __launch_bounds__(128) void colsum_kernel(const float* __restrict__ r,
                                                     float* __restrict__ cs, int Nn) {
  int col = threadIdx.x;
  int row0 = blockIdx.x * 256;
  int row1 = row0 + 256; if (row1 > Nn) row1 = Nn;
  float s = 0.f;
  for (int row = row0; row < row1; ++row) s += r[(size_t)row * HIDW + col];
  atomicAdd(&cs[col], s);
}

__global__ __launch_bounds__(128) void colvar_kernel(const float* __restrict__ r,
                                                     const float* __restrict__ ms,
                                                     const float* __restrict__ cs,
                                                     float* __restrict__ c2, int Nn) {
  int col = threadIdx.x;
  float m2 = ms[col] * (cs[col] / (float)Nn);
  int row0 = blockIdx.x * 256;
  int row1 = row0 + 256; if (row1 > Nn) row1 = Nn;
  float s = 0.f;
  for (int row = row0; row < row1; ++row) { float d = r[(size_t)row * HIDW + col] - m2; s += d * d; }
  atomicAdd(&c2[col], s);
}

__global__ __launch_bounds__(256) void gn_final_kernel(
    const float* __restrict__ r, const float* __restrict__ ms,
    const float* __restrict__ cs, const float* __restrict__ c2,
    const float* __restrict__ g, const float* __restrict__ b,
    float* __restrict__ h, int Nn) {
  long long i = (long long)blockIdx.x * 256 + threadIdx.x;
  if (i >= (long long)Nn * HIDW) return;
  int col = (int)(i & 127);
  float mu  = cs[col] / (float)Nn;
  float var = c2[col] / (float)Nn;
  float c = r[i] - ms[col] * mu;
  h[i] = gelu_exact(g[col] * c * rsqrtf(var + 1e-5f) + b[col]);
}

// ---- linear edge-to-edge attention: KV reduction + apply ----
#define KV_EPB 1024
__global__ __launch_bounds__(256) void kv_reduce_kernel(
    const float* __restrict__ ek, const float* __restrict__ ev,
    float* __restrict__ kvg, float* __restrict__ ksg, int E) {
  __shared__ float skv[2048];
  __shared__ float sks[128];
  __shared__ float ebuf[8 * 128];
  __shared__ float vbuf[8 * 128];
  int tid = threadIdx.x;
  for (int i = tid; i < 2048; i += 256) skv[i] = 0.f;
  if (tid < 128) sks[tid] = 0.f;
  __syncthreads();
  int e0 = blockIdx.x * KV_EPB;
  int e1 = e0 + KV_EPB; if (e1 > E) e1 = E;
  for (int base = e0; base < e1; base += 8) {
    int cnt = e1 - base; if (cnt > 8) cnt = 8;
    for (int i = tid; i < cnt * 128; i += 256) {
      ebuf[i] = ek[(size_t)base * HIDW + i];
      vbuf[i] = ev[(size_t)base * HIDW + i];
    }
    __syncthreads();
#pragma unroll
    for (int j = 0; j < 8; ++j) {
      int cell = tid + j * 256;
      int hh = cell >> 8, dd = (cell >> 4) & 15, mm = cell & 15;
      float s = skv[cell];
      for (int ii = 0; ii < cnt; ++ii)
        s += ebuf[ii * 128 + hh * HDIM + dd] * vbuf[ii * 128 + hh * HDIM + mm];
      skv[cell] = s;
    }
    if (tid < 128) {
      float s = sks[tid];
      for (int ii = 0; ii < cnt; ++ii) s += ebuf[ii * 128 + tid];
      sks[tid] = s;
    }
    __syncthreads();
  }
  for (int i = tid; i < 2048; i += 256) atomicAdd(&kvg[i], skv[i]);
  if (tid < 128) atomicAdd(&ksg[tid], sks[tid]);
}

__global__ __launch_bounds__(256) void linattn_apply_kernel(
    const float* __restrict__ eq, const float* __restrict__ kvg,
    const float* __restrict__ ksg, float* __restrict__ o, int E) {
  __shared__ float skv[2048];
  __shared__ float sks[128];
  int tid = threadIdx.x;
  for (int i = tid; i < 2048; i += 256) skv[i] = kvg[i];
  if (tid < 128) sks[tid] = ksg[tid];
  __syncthreads();
  int idx = blockIdx.x * 256 + tid;
  if (idx >= E * NHEADS) return;
  int e = idx >> 3, hh = idx & 7;
  const float* qp = eq + (size_t)e * HIDW + hh * HDIM;
  float qv[HDIM];
#pragma unroll
  for (int i = 0; i < HDIM; ++i) qv[i] = qp[i];
  float z = 1e-6f;
#pragma unroll
  for (int d = 0; d < HDIM; ++d) z += qv[d] * sks[hh * HDIM + d];
  float* op = o + (size_t)e * HIDW + hh * HDIM;
#pragma unroll
  for (int mm = 0; mm < HDIM; ++mm) {
    float num = 0.f;
#pragma unroll
    for (int d = 0; d < HDIM; ++d) num += qv[d] * skv[hh * 256 + d * 16 + mm];
    op[mm] = num / z;
  }
}

// ---------------------------------------------------------------------------
// Orchestration
// ---------------------------------------------------------------------------
extern "C" void kernel_launch(void* const* d_in, const int* in_sizes, int n_in,
                              void* d_out, int out_size, void* d_ws, size_t ws_size,
                              hipStream_t stream) {
  (void)n_in; (void)out_size; (void)ws_size;
  const int N = in_sizes[0] / 64;
  const int E = in_sizes[2] / 32;

  auto F = [&](int i) { return (const float*)d_in[i]; };
  const float* x     = F(0);
  const int*   ei    = (const int*)d_in[1];
  const int*   src   = ei;
  const int*   dst   = ei + E;
  const float* eatt  = F(2);
  // leaf indices: 3 ne.w, 4 ne.b, 5/6 node_ln, 7 ee.w, 8 ee.b, 9/10 edge_ln
  const int LBASE = 11, LSTRIDE = 29;

  // ---- workspace carve ----
  char* wsp = (char*)d_ws;
  size_t off = 0;
  auto carve = [&](size_t bytes) {
    void* p = wsp + off;
    off = (off + bytes + 255) & ~(size_t)255;
    return p;
  };
  unsigned short* wb_ne = (unsigned short*)carve(8192 * 2);
  unsigned short* wb_ee = (unsigned short*)carve(4096 * 2);
  struct LayerW { unsigned short *wq, *wk, *wv, *we, *nao, *mlp1, *mlp2, *gate, *weq, *wek, *wev, *weo; };
  LayerW lw[2];
  for (int l = 0; l < 2; ++l) {
    lw[l].wq   = (unsigned short*)carve(16384 * 2);
    lw[l].wk   = (unsigned short*)carve(16384 * 2);
    lw[l].wv   = (unsigned short*)carve(16384 * 2);
    lw[l].we   = (unsigned short*)carve(16384 * 2);
    lw[l].nao  = (unsigned short*)carve(16384 * 2);
    lw[l].mlp1 = (unsigned short*)carve(49152 * 2);
    lw[l].mlp2 = (unsigned short*)carve(16384 * 2);
    lw[l].gate = (unsigned short*)carve(49152 * 2);
    lw[l].weq  = (unsigned short*)carve(16384 * 2);
    lw[l].wek  = (unsigned short*)carve(16384 * 2);
    lw[l].wev  = (unsigned short*)carve(16384 * 2);
    lw[l].weo  = (unsigned short*)carve(16384 * 2);
  }
  float* bufE1  = (float*)carve((size_t)E * HIDW * 4);
  float* bufE2  = (float*)carve((size_t)E * HIDW * 4);
  float* bufE3  = (float*)carve((size_t)E * HIDW * 4);
  float* segmax = (float*)carve((size_t)N * NHEADS * 4);
  float* segsum = (float*)carve((size_t)N * NHEADS * 4);
  float* cs     = (float*)carve(128 * 4);
  float* c2     = (float*)carve(128 * 4);
  float* kvg    = (float*)carve(2048 * 4);
  float* ksg    = (float*)carve(128 * 4);

  float* h  = (float*)d_out;
  float* ea = (float*)d_out + (size_t)N * HIDW;

  // node-sized scratch inside bufE3 (only live during node attention stage)
  float* qb   = bufE3;
  float* kb   = bufE3 + (size_t)1 * N * HIDW;
  float* vb   = bufE3 + (size_t)2 * N * HIDW;
  float* aggb = bufE3 + (size_t)3 * N * HIDW;
  float* rb   = bufE3 + (size_t)4 * N * HIDW;

  auto conv = [&](const float* s, unsigned short* d, int n) {
    f2bf_kernel<<<dim3((n + 255) / 256), dim3(256), 0, stream>>>(s, d, n);
  };
  auto gemm = [&](const float* A0, const int* R0, int W0,
                  const float* A1, const int* R1, int W1,
                  const float* A2, const int* R2, int W2,
                  const unsigned short* Wt, const float* bias, float* Y, int M, int K) {
    wmma_gemm_kernel<<<dim3((M + 31) / 32), dim3(256), 0, stream>>>(
        A0, R0, W0, A1, R1, W1, A2, R2, W2, Wt, bias, Y, M, K);
  };
  auto gemm1 = [&](const float* A, const unsigned short* Wt, const float* bias,
                   float* Y, int M, int K) {
    gemm(A, nullptr, K, nullptr, nullptr, 0, nullptr, nullptr, 0, Wt, bias, Y, M, K);
  };
  auto lnf = [&](const float* A, const float* B, const float* C, const float* g,
                 const float* b, float* o, int R, int mode, int gl) {
    ln_fused_kernel<<<dim3((R + 7) / 8), dim3(256), 0, stream>>>(A, B, C, g, b, o, R, mode, gl);
  };
  auto fill = [&](float* p, float v, long long n) {
    fill_kernel<<<dim3((unsigned)((n + 255) / 256)), dim3(256), 0, stream>>>(p, v, n);
  };

  // ---- weight conversions (fp32 -> bf16 bits) ----
  conv(F(3), wb_ne, 8192);
  conv(F(7), wb_ee, 4096);
  for (int l = 0; l < 2; ++l) {
    int p = LBASE + l * LSTRIDE;
    conv(F(p + 0),  lw[l].wq,   16384);
    conv(F(p + 1),  lw[l].wk,   16384);
    conv(F(p + 2),  lw[l].wv,   16384);
    conv(F(p + 3),  lw[l].we,   16384);
    conv(F(p + 4),  lw[l].nao,  16384);
    conv(F(p + 9),  lw[l].mlp1, 49152);
    conv(F(p + 13), lw[l].mlp2, 16384);
    conv(F(p + 15), lw[l].gate, 49152);
    conv(F(p + 19), lw[l].weq,  16384);
    conv(F(p + 21), lw[l].wek,  16384);
    conv(F(p + 23), lw[l].wev,  16384);
    conv(F(p + 25), lw[l].weo,  16384);
  }

  const unsigned tE8   = (unsigned)((E * NHEADS + 255) / 256);
  const unsigned tN128 = (unsigned)(((long long)N * HIDW + 255) / 256);
  const unsigned tE128 = (unsigned)(((long long)E * HIDW + 255) / 256);

  // ---- encoders ----
  gemm1(x, wb_ne, F(4), bufE1, N, 64);
  lnf(bufE1, nullptr, nullptr, F(5), F(6), h, N, 0, 0);
  gemm1(eatt, wb_ee, F(8), bufE1, E, 32);
  lnf(bufE1, nullptr, nullptr, F(9), F(10), ea, E, 0, 0);

  for (int l = 0; l < 2; ++l) {
    int p = LBASE + l * LSTRIDE;

    // ---- EdgeAugmentedAttention ----
    gemm1(h,  lw[l].wq, nullptr, qb, N, 128);
    gemm1(h,  lw[l].wk, nullptr, kb, N, 128);
    gemm1(h,  lw[l].wv, nullptr, vb, N, 128);
    gemm1(ea, lw[l].we, nullptr, bufE1, E, 128);          // eemb
    fill(segmax, -3.0e38f, (long long)N * NHEADS);
    fill(segsum, 0.f, (long long)N * NHEADS);
    fill(aggb, 0.f, (long long)N * HIDW);
    attn_score_kernel<<<tE8, 256, 0, stream>>>(qb, kb, bufE1, src, dst, bufE2, segmax, E);
    attn_exp_kernel<<<tE8, 256, 0, stream>>>(bufE2, dst, segmax, segsum, E);
    attn_agg_kernel<<<tE8, 256, 0, stream>>>(bufE2, src, dst, segsum, vb, bufE1, aggb, E);
    gemm1(aggb, lw[l].nao, F(p + 5), rb, N, 128);
    add2_kernel<<<tN128, 256, 0, stream>>>(rb, h, rb, (long long)N * HIDW);
    fill(cs, 0.f, 128);
    fill(c2, 0.f, 128);
    colsum_kernel<<<dim3((N + 255) / 256), dim3(128), 0, stream>>>(rb, cs, N);
    colvar_kernel<<<dim3((N + 255) / 256), dim3(128), 0, stream>>>(rb, F(p + 8), cs, c2, N);
    gn_final_kernel<<<tN128, 256, 0, stream>>>(rb, F(p + 8), cs, c2, F(p + 6), F(p + 7), h, N);

    // ---- EdgeUpdaterModule (concat fused into GEMM gather, K=384) ----
    gemm(h, src, 128, h, dst, 128, ea, nullptr, 128, lw[l].mlp1, F(p + 10), bufE1, E, 384);
    gemm(h, src, 128, h, dst, 128, ea, nullptr, 128, lw[l].gate, F(p + 16), bufE2, E, 384);
    lnf(bufE1, nullptr, nullptr, F(p + 11), F(p + 12), bufE1, E, 0, 1);   // gelu(ln(u))
    gemm1(bufE1, lw[l].mlp2, F(p + 14), bufE3, E, 128);                   // u2
    lnf(bufE3, bufE2, ea, F(p + 17), F(p + 18), ea, E, 1, 0);             // ln(u2*sig(gate)+ea)

    // ---- FastEdgeToEdgeAttention ----
    gemm1(ea, lw[l].weq, F(p + 20), bufE1, E, 128);
    gemm1(ea, lw[l].wek, F(p + 22), bufE2, E, 128);
    gemm1(ea, lw[l].wev, F(p + 24), bufE3, E, 128);
    elu1_kernel<<<tE128, 256, 0, stream>>>(bufE1, (long long)E * HIDW);
    elu1_kernel<<<tE128, 256, 0, stream>>>(bufE2, (long long)E * HIDW);
    fill(kvg, 0.f, 2048);
    fill(ksg, 0.f, 128);
    kv_reduce_kernel<<<dim3((E + KV_EPB - 1) / KV_EPB), dim3(256), 0, stream>>>(bufE2, bufE3, kvg, ksg, E);
    linattn_apply_kernel<<<tE8, 256, 0, stream>>>(bufE1, kvg, ksg, bufE2, E);
    gemm1(bufE2, lw[l].weo, F(p + 26), bufE3, E, 128);
    lnf(bufE3, nullptr, ea, F(p + 27), F(p + 28), ea, E, 2, 0);           // ln(ea+gelu(o))
  }
}
